// parallel_RNN_44555990729257
// MI455X (gfx1250) — matrix operand order repaired and checked by reading, not compiled, not run on hardware
//
#include <hip/hip_runtime.h>
#include <math.h>

typedef __attribute__((ext_vector_type(16))) _Float16 v16h;
typedef __attribute__((ext_vector_type(8)))  _Float16 v8h;
typedef __attribute__((ext_vector_type(8)))  float    v8f;

#define VSZ     64
#define HSZ     64
#define FULLW   128
#define NWIN    256
#define TSTEPS  64
#define RSTRIDE 136   // 128 + 8 halves: 272B row pitch -> 4-bank skew per row, conflict-free A loads

__device__ __forceinline__ _Float16 f2h(float f) { return (_Float16)f; } // v_cvt_f16_f32 (RNE)

__global__ __launch_bounds__(256, 1)
void prnn_kernel(const float* __restrict__ X,
                 const float* __restrict__ W,
                 const float* __restrict__ b,
                 float* __restrict__ out) {
  // Double-buffered recurrent state, f16: 2 * 64 * 136 * 2B = 34,816 B of LDS
  __shared__ _Float16 R[2][VSZ * RSTRIDE];

  const int win  = blockIdx.x;
  const int tid  = threadIdx.x;
  const int wave = tid >> 5;
  const int lane = tid & 31;
  const int lmod = lane & 15;
  const bool hi  = lane >= 16;

  // Tile assignment: wave -> 16-row block m0, and one 64-col half (4 n-tiles)
  const int m0    = (wave >> 1) * 16;
  const int nhalf = wave & 1;          // 0: visible cols 0..63, 1: hidden cols 64..127
  const int nbase = nhalf * VSZ;

  // R0 = 0
  for (int i = tid; i < VSZ * RSTRIDE; i += 256) {
    R[0][i] = (_Float16)0.0f;
    R[1][i] = (_Float16)0.0f;
  }

  // ---- Preload W^T as WMMA B-fragments into registers (constant across time) ----
  // B[k][n] = W[n][k].  B layout (16-bit, 32x16), matching A: lanes 0-15 hold K=0..7 & 16..23 for
  // N=lane; lanes 16-31 hold K=8..15 & 24..31 for N=lane-16.
  v16h Bfrag[4][4];   // [n-tile][k-block]  => 128 VGPRs, stays register-resident
  #pragma unroll
  for (int nt = 0; nt < 4; ++nt) {
    const int n = nbase + nt * 16 + lmod;
    #pragma unroll
    for (int kb = 0; kb < 4; ++kb) {
      const float* wp = W + n * FULLW + kb * 32 + (hi ? 8 : 0);
      v16h f;
      #pragma unroll
      for (int i = 0; i < 16; ++i) f[i] = f2h(wp[i < 8 ? i : i + 8]);
      Bfrag[nt][kb] = f;
    }
  }

  float bias[4];
  #pragma unroll
  for (int nt = 0; nt < 4; ++nt) bias[nt] = b[nbase + nt * 16 + lmod];

  __syncthreads();

  int p = 0;
  for (int t = 0; t < TSTEPS; ++t) {
    // Input slice for visible columns (broadcast down rows in the reference)
    float xv[4] = {0.f, 0.f, 0.f, 0.f};
    const float* xp = X + (size_t)t * (NWIN * VSZ) + win * VSZ;
    if (nhalf == 0) {
      #pragma unroll
      for (int nt = 0; nt < 4; ++nt) xv[nt] = xp[nt * 16 + lmod];
      if (t + 1 < TSTEPS) __builtin_prefetch(xp + NWIN * VSZ, 0, 0); // global_prefetch
    }

    // ---- Preload ALL four A fragments (distinct regs) so the 8 ds_load_b128
    // issue together, one dscnt wait, then an uninterrupted 16-WMMA burst with
    // four independent accumulator chains (no per-k-block WAR stalls).
    // ISA A layout (16-bit, 16x32): lanes 0-15: K=0..7 (v0-3) & 16..23 (v4-7);
    // lanes 16-31: K=8..15 & 24..31. Two contiguous 16B LDS reads per fragment.
    v16h Afrag[4];
    #pragma unroll
    for (int kb = 0; kb < 4; ++kb) {
      const _Float16* rp = &R[p][(m0 + lmod) * RSTRIDE + kb * 32 + (hi ? 8 : 0)];
      v8h a_lo = *reinterpret_cast<const v8h*>(rp);
      v8h a_hi = *reinterpret_cast<const v8h*>(rp + 16);
      Afrag[kb] = __builtin_shufflevector(a_lo, a_hi,
                                          0, 1, 2, 3, 4, 5, 6, 7,
                                          8, 9, 10, 11, 12, 13, 14, 15);
    }

    v8f acc[4] = { {}, {}, {}, {} };

    // ---- GEMM: WR(m0..m0+15, nbase..nbase+63) = R * W^T ----
    #pragma unroll
    for (int kb = 0; kb < 4; ++kb) {
      #pragma unroll
      for (int nt = 0; nt < 4; ++nt) {
        acc[nt] = __builtin_amdgcn_wmma_f32_16x16x32_f16(
            false, Afrag[kb], false, Bfrag[nt][kb], (short)0, acc[nt], false, false);
      }
    }

    // ---- Epilogue: U = Lam*(WR+b) with off-diag visible <- x_t; R_next = tanh(U) ----
    // C/D layout: VGPR r, lane<16 -> (M=m0+r, N=n); lane>=16 -> (M=m0+8+r, N=n)
    _Float16* Rn = R[p ^ 1];
    #pragma unroll
    for (int nt = 0; nt < 4; ++nt) {
      const int n = nbase + nt * 16 + lmod;
      const float bia = bias[nt];
      float tx = 0.0f;
      if (nhalf == 0) tx = tanhf(xv[nt]);
      #pragma unroll
      for (int r = 0; r < 8; ++r) {
        const int m = m0 + r + (hi ? 8 : 0);
        const float val = acc[nt][r] + bia;
        float nv;
        if (nhalf != 0) {              // hidden columns: keep WR+b
          nv = tanhf(val);
        } else if (n == m) {           // visible diagonal: keep WR+b; output is pre-tanh U
          nv = tanhf(val);
          if (t == TSTEPS - 1) out[win * VSZ + m] = val;
        } else {                       // visible off-diagonal: tiled input
          nv = tx;
        }
        Rn[m * RSTRIDE + n] = f2h(nv);
      }
    }

    __syncthreads();
    p ^= 1;
  }
}

extern "C" void kernel_launch(void* const* d_in, const int* in_sizes, int n_in,
                              void* d_out, int out_size, void* d_ws, size_t ws_size,
                              hipStream_t stream) {
  const float* X = (const float*)d_in[0];   // [T, NW*VS] f32
  const float* W = (const float*)d_in[1];   // [FULL, FULL] f32
  const float* b = (const float*)d_in[2];   // [FULL] f32
  float* out = (float*)d_out;               // [NW*VS] f32

  prnn_kernel<<<NWIN, 256, 0, stream>>>(X, W, b, out);
}
